// GCN_51582557225371
// MI455X (gfx1250) — matrix operand order repaired and checked by reading, not compiled, not run on hardware
//
#include <hip/hip_runtime.h>

typedef __bf16 bf16;
typedef __attribute__((ext_vector_type(16))) __bf16 v16bf;
typedef __attribute__((ext_vector_type(8)))  __bf16 v8bf;
typedef __attribute__((ext_vector_type(8)))  float  v8f;
typedef __attribute__((ext_vector_type(4)))  unsigned int u32x4;
typedef __attribute__((ext_vector_type(4)))  float  f32x4;

#define NROWS 12000
#define EPAD  512
#define KADJ  12000
#define H4LD  32
#define H4ROWS 12032   // 94*128, padded for decode edge tiles

// ---- helpers -------------------------------------------------------------

__device__ __forceinline__ bf16 f2bf(float f) {
  union { float f; unsigned int u; } a; a.f = f;
  unsigned int u = a.u + 0x7FFFu + ((a.u >> 16) & 1u);   // round-to-nearest-even
  union { unsigned short s; bf16 b; } r; r.s = (unsigned short)(u >> 16);
  return r.b;
}

union V16 { v16bf v; u32x4 q[2]; };
union V8  { v8bf  v; u32x4 q;   };

__device__ __forceinline__ v8f wmma_bf16(v16bf a, v16bf b, v8f c) {
  // D = A(16x32 bf16) * B(32x16 bf16) + C(16x16 f32)
  return __builtin_amdgcn_wmma_f32_16x16x32_bf16(false, a, false, b, (short)0, c,
                                                 false, false);
}

__device__ __forceinline__ v8f vzero8() {
  v8f z = {0.f,0.f,0.f,0.f,0.f,0.f,0.f,0.f};
  return z;
}

// ---- conversion kernels --------------------------------------------------

// fp32 [rows][cols_in] -> bf16 [rows][ld], zero-padded columns
__global__ __launch_bounds__(256) void k_conv_pad(const float* __restrict__ src,
                                                  bf16* __restrict__ dst,
                                                  int rows, int cols_in, int ld) {
  int idx = blockIdx.x * 256 + threadIdx.x;
  if (idx >= rows * ld) return;
  int r = idx / ld, c = idx - r * ld;
  float v = (c < cols_in) ? src[(size_t)r * cols_in + c] : 0.f;
  dst[idx] = f2bf(v);
}

// W fp32 [Kw][Nw] -> WT bf16 [512][512] (row = n, col = k), zero-padded
__global__ __launch_bounds__(256) void k_conv_wt(const float* __restrict__ W,
                                                 bf16* __restrict__ WT,
                                                 int Kw, int Nw) {
  int idx = blockIdx.x * 256 + threadIdx.x;
  if (idx >= EPAD * EPAD) return;
  int n = idx >> 9, k = idx & 511;
  float v = (k < Kw && n < Nw) ? W[(size_t)k * Nw + n] : 0.f;
  WT[idx] = f2bf(v);
}

// ---- kernel 2: ZT = (H @ W)^T  (bf16 in, bf16 transposed out) -----------
// H: [NROWS][EPAD] bf16 row-major ; WT: [EPAD][EPAD] (row n, col k)
// ZT: [EPAD][NROWS] bf16 (row n, col m)

__global__ __launch_bounds__(256) void k_hw_gemm(const bf16* __restrict__ H,
                                                 const bf16* __restrict__ WT,
                                                 bf16* __restrict__ ZT) {
  const int m0   = blockIdx.x * 32;
  const int lane = threadIdx.x & 31;
  const int w    = threadIdx.x >> 5;
  const int n0   = w * 64;
  const int lh   = lane >> 4;       // 0 or 1
  const int ll   = lane & 15;

  v8f acc[2][4];
  for (int mt = 0; mt < 2; ++mt)
    for (int nt = 0; nt < 4; ++nt) acc[mt][nt] = vzero8();

  for (int k0 = 0; k0 < EPAD; k0 += 32) {
    V16 a[2];
    for (int mt = 0; mt < 2; ++mt) {
      int m  = m0 + mt * 16 + ll;
      int ko = k0 + lh * 8;
      a[mt].q[0] = *(const u32x4*)(H + (size_t)m * EPAD + ko);
      a[mt].q[1] = *(const u32x4*)(H + (size_t)m * EPAD + ko + 16);
    }
    for (int nt = 0; nt < 4; ++nt) {
      int n  = n0 + nt * 16 + ll;
      int kb = k0 + lh * 8;
      V16 b;
      const u32x4* pb = (const u32x4*)(WT + (size_t)n * EPAD + kb);
      b.q[0] = pb[0]; b.q[1] = pb[2];
      acc[0][nt] = wmma_bf16(a[0].v, b.v, acc[0][nt]);
      acc[1][nt] = wmma_bf16(a[1].v, b.v, acc[1][nt]);
    }
  }

  // transposed packed store: per lane, 8 consecutive m for fixed n
  for (int mt = 0; mt < 2; ++mt)
    for (int nt = 0; nt < 4; ++nt) {
      int n  = n0 + nt * 16 + ll;
      int mb = m0 + mt * 16 + lh * 8;
      V8 s;
      for (int r = 0; r < 8; ++r) s.v[r] = f2bf(acc[mt][nt][r]);
      *(u32x4*)(ZT + (size_t)n * NROWS + mb) = s.q;
    }
}

// ---- kernel 1: H = relu(adj @ Z)  (adj fp32 -> bf16 on the fly) ---------
// Double-buffered LDS software pipeline: one barrier per K-step, the global
// load for tile kk+2 is issued one full compute iteration ahead.
// adj: [NROWS][KADJ] fp32 ; ZT: [EPAD][KADJ] bf16 (row n, col k)
// Hout bf16 [NROWS][ldh]; optional Fout fp32 [NROWS][ldf] (layer-4 h output)

__global__ __launch_bounds__(256) void k_adj_gemm(const float* __restrict__ adj,
                                                  const bf16* __restrict__ ZT,
                                                  bf16* __restrict__ Hout, int ldh,
                                                  float* __restrict__ Fout, int ldf,
                                                  int n_f32, int n_compute,
                                                  int n_store) {
  __shared__ bf16 As[2][32][40];  // double buffer; row padded to 80B vs bank conflicts

  const int m0   = blockIdx.x * 32;
  const int lane = threadIdx.x & 31;
  const int w    = threadIdx.x >> 5;
  const int n0   = w * 64;
  const int lh   = lane >> 4;
  const int ll   = lane & 15;

  const int row = threadIdx.x >> 3;         // 0..31
  const int cg  = (threadIdx.x & 7) * 4;    // 0..28
  const float* arow = adj + (size_t)(m0 + row) * KADJ + cg;

  v8f acc[2][4];
  for (int mt = 0; mt < 2; ++mt)
    for (int nt = 0; nt < 4; ++nt) acc[mt][nt] = vzero8();

  const int niter = KADJ / 32;              // 375

  // prologue: tile 0 -> LDS[0]; issue load of tile 1
  {
    f32x4 f0 = *(const f32x4*)(arow);
    bf16* d = &As[0][row][cg];
    d[0] = f2bf(f0.x); d[1] = f2bf(f0.y); d[2] = f2bf(f0.z); d[3] = f2bf(f0.w);
  }
  f32x4 fnext = *(const f32x4*)(arow + 32); // tile 1 (niter >= 2 always)
  __syncthreads();

  for (int kk = 0; kk < niter; ++kk) {
    const int cur = kk & 1;

    // stage tile kk+1 into the other buffer (readers of it finished at the
    // barrier ending iteration kk-1), then issue the load for tile kk+2
    if (kk + 1 < niter) {
      bf16* d = &As[1 - cur][row][cg];
      d[0] = f2bf(fnext.x); d[1] = f2bf(fnext.y);
      d[2] = f2bf(fnext.z); d[3] = f2bf(fnext.w);
    }
    if (kk + 2 < niter) {
      fnext = *(const f32x4*)(arow + (size_t)(kk + 2) * 32);
    }

    if (n0 < n_compute) {
      const int k0   = kk * 32;
      const int koff = lh * 8;
      V16 a[2];
      for (int mt = 0; mt < 2; ++mt) {
        a[mt].q[0] = *(const u32x4*)&As[cur][mt * 16 + ll][koff];
        a[mt].q[1] = *(const u32x4*)&As[cur][mt * 16 + ll][16 + koff];
      }
      for (int nt = 0; nt < 4; ++nt) {
        int nb = n0 + nt * 16;
        if (nb >= n_compute) break;
        int n  = nb + ll;
        int kb = k0 + lh * 8;
        V16 b;
        const u32x4* pb = (const u32x4*)(ZT + (size_t)n * KADJ + kb);
        b.q[0] = pb[0]; b.q[1] = pb[2];
        acc[0][nt] = wmma_bf16(a[0].v, b.v, acc[0][nt]);
        acc[1][nt] = wmma_bf16(a[1].v, b.v, acc[1][nt]);
      }
    }
    __syncthreads();
  }

  // epilogue: relu, bf16 store (+ optional fp32 store for layer-4 h)
  for (int mt = 0; mt < 2; ++mt)
    for (int nt = 0; nt < 4; ++nt) {
      int n  = n0 + nt * 16 + ll;
      int mb = m0 + mt * 16 + lh * 8;
      for (int r = 0; r < 8; ++r) {
        float v = acc[mt][nt][r];
        v = v > 0.f ? v : 0.f;
        int m = mb + r;
        if (n < n_store) Hout[(size_t)m * ldh + n] = f2bf(v);
        if (Fout && n < n_f32) Fout[(size_t)m * ldf + n] = v;
      }
    }
}

// ---- kernel 3: A_pred = sigmoid(H4 @ H4^T), K = 32 (padded) -------------
// Hp: [H4ROWS][32] bf16, rows >= 12000 and cols >= 10 are zero.

__global__ __launch_bounds__(256) void k_decode(const bf16* __restrict__ Hp,
                                                float* __restrict__ out) {
  const int bm   = blockIdx.x * 128;
  const int bn   = blockIdx.y * 128;
  const int lane = threadIdx.x & 31;
  const int w    = threadIdx.x >> 5;
  const int wm   = (w & 1) * 64;
  const int wn   = (w >> 1) * 32;
  const int lh   = lane >> 4;
  const int ll   = lane & 15;

  V16 a[4];
  const int koff = lh * 8;
  for (int mt = 0; mt < 4; ++mt) {
    int m = bm + wm + mt * 16 + ll;
    a[mt].q[0] = *(const u32x4*)(Hp + (size_t)m * H4LD + koff);
    a[mt].q[1] = *(const u32x4*)(Hp + (size_t)m * H4LD + 16 + koff);
  }
  V16 b[2];
  const int kb = lh * 8;
  for (int nt = 0; nt < 2; ++nt) {
    int n = bn + wn + nt * 16 + ll;
    const u32x4* pb = (const u32x4*)(Hp + (size_t)n * H4LD + kb);
    b[nt].q[0] = pb[0]; b[nt].q[1] = pb[2];
  }

  v8f acc[4][2];
  for (int mt = 0; mt < 4; ++mt)
    for (int nt = 0; nt < 2; ++nt)
      acc[mt][nt] = wmma_bf16(a[mt].v, b[nt].v, vzero8());

  for (int mt = 0; mt < 4; ++mt)
    for (int nt = 0; nt < 2; ++nt) {
      int n  = bn + wn + nt * 16 + ll;
      int mb = bm + wm + mt * 16 + lh * 8;
      for (int r = 0; r < 8; ++r) {
        int m = mb + r;
        if (m < NROWS && n < NROWS) {
          float v = acc[mt][nt][r];
          out[(size_t)m * NROWS + n] = 1.f / (1.f + __expf(-v));
        }
      }
    }
}

// ---- host orchestration --------------------------------------------------

extern "C" void kernel_launch(void* const* d_in, const int* in_sizes, int n_in,
                              void* d_out, int out_size, void* d_ws, size_t ws_size,
                              hipStream_t stream) {
  const float* x   = (const float*)d_in[0];
  const float* adj = (const float*)d_in[1];
  const float* Wl[4] = {(const float*)d_in[2], (const float*)d_in[3],
                        (const float*)d_in[4], (const float*)d_in[5]};

  float* Apred = (float*)d_out;
  float* hout  = Apred + (size_t)NROWS * NROWS;   // [12000][10] fp32

  bf16* Ha = (bf16*)d_ws;                         // [12000][512]
  bf16* Hb = Ha + (size_t)NROWS * EPAD;           // [12000][512]
  bf16* ZT = Hb + (size_t)NROWS * EPAD;           // [512][12000]
  bf16* WT = ZT + (size_t)EPAD * NROWS;           // [512][512]
  bf16* H4 = WT + (size_t)EPAD * EPAD;            // [12032][32]

  // H0 = bf16(x), zero-padded to 512 columns
  {
    int total = NROWS * EPAD;
    k_conv_pad<<<(total + 255) / 256, 256, 0, stream>>>(x, Ha, NROWS, 500, EPAD);
  }

  const int Ein[4]  = {500, 500, 500, 500};
  const int Eout[4] = {500, 500, 500, 10};
  const bf16* Hprev = Ha;
  bf16* Hnext = Hb;

  for (int l = 0; l < 4; ++l) {
    k_conv_wt<<<(EPAD * EPAD + 255) / 256, 256, 0, stream>>>(Wl[l], WT, Ein[l], Eout[l]);
    k_hw_gemm<<<NROWS / 32, 256, 0, stream>>>(Hprev, WT, ZT);
    if (l < 3) {
      k_adj_gemm<<<NROWS / 32, 256, 0, stream>>>(adj, ZT, Hnext, EPAD,
                                                 nullptr, 0, 0,
                                                 /*n_compute=*/EPAD, /*n_store=*/EPAD);
      const bf16* t = Hprev; Hprev = Hnext; Hnext = (bf16*)t;
    } else {
      hipMemsetAsync(H4, 0, (size_t)H4ROWS * H4LD * sizeof(bf16), stream);
      k_adj_gemm<<<NROWS / 32, 256, 0, stream>>>(adj, ZT, H4, H4LD,
                                                 hout, 10, /*n_f32=*/10,
                                                 /*n_compute=*/16, /*n_store=*/16);
    }
  }

  dim3 grid((NROWS + 127) / 128, (NROWS + 127) / 128);
  k_decode<<<grid, 256, 0, stream>>>(H4, Apred);
}
